// Encoder_With_Codebook_65034394796699
// MI455X (gfx1250) — matrix operand order repaired and checked by reading, not compiled, not run on hardware
//
#include <hip/hip_runtime.h>
#include <hip/hip_bf16.h>
#include <math.h>

typedef __attribute__((ext_vector_type(16))) __bf16 v16bf;
typedef __attribute__((ext_vector_type(8)))  __bf16 v8bf;
typedef __attribute__((ext_vector_type(8)))  float  v8f;

// ---------------- helpers ----------------
static __device__ __forceinline__ unsigned short f32_to_bf16(float f) {
    union { float f; unsigned u; } v; v.f = f;
    unsigned u = v.u;
    unsigned rnd = 0x7FFFu + ((u >> 16) & 1u);
    return (unsigned short)((u + rnd) >> 16);
}

#define BF16CAT(lo, hi) __builtin_shufflevector(lo, hi, 0,1,2,3,4,5,6,7,8,9,10,11,12,13,14,15)

// ---------------- WMMA GEMM: C(MxN) = A(MxK,bf16) * Bt(NxK,bf16)^T + bias ----------------
// Software pipelined via 2x-unrolled buffer rotation (no register copies):
//   loads->buf1 ; 4x wmma(buf0) ; loads->buf0 ; 4x wmma(buf1)
// Requires Kk % 64 == 0 (K dims padded accordingly).
// act: 0=none, 1=relu, 2=exp-clamp(1e-6)  (variance head)
__global__ __launch_bounds__(256) void gemm_bf16_wmma(
    const __bf16* __restrict__ A, int lda,
    const __bf16* __restrict__ Bt, int ldbt,
    const float* __restrict__ bias,
    float* __restrict__ C, int ldc,
    int M, int Nn, int Kk, int act)
{
    const int lane = threadIdx.x & 31;
    const int wid  = blockIdx.x * (blockDim.x >> 5) + (threadIdx.x >> 5);
    const int tilesN = Nn >> 6;                 // each wave covers 64 output columns
    const int tiles  = (M >> 4) * tilesN;
    if (wid >= tiles) return;                   // wave-uniform exit (EXEC all-ones for WMMA)
    const int tm = (wid / tilesN) << 4;
    const int tn = (wid % tilesN) << 6;
    const int r  = lane & 15;
    const int hi = lane >> 4;

    v8f acc[4] = {};
    const int aoff = hi << 4;                   // A: elems 0..15 -> K=k0+hi*16+e contiguous
    const __bf16* Arow = A + (size_t)(tm + r) * lda;
    const __bf16* Brow[4];
    #pragma unroll
    for (int t = 0; t < 4; ++t)                 // B: lane column n=r, K=k0+hi*16+e contiguous
        Brow[t] = Bt + (size_t)(tn + t * 16 + r) * ldbt + (hi << 4);

    // prologue: buffer0 <- k = 0
    v16bf a0b = BF16CAT(*(const v8bf*)(Arow + aoff), *(const v8bf*)(Arow + 8 + aoff));
    v16bf b0b[4];
    #pragma unroll
    for (int t = 0; t < 4; ++t)
        b0b[t] = BF16CAT(*(const v8bf*)(Brow[t]), *(const v8bf*)(Brow[t] + 8));

    for (int k0 = 0; k0 < Kk; k0 += 64) {
        const int k1 = k0 + 32;                               // always < Kk (Kk % 64 == 0)
        // prefetch buffer1 <- k1
        v16bf a1b = BF16CAT(*(const v8bf*)(Arow + k1 + aoff),
                            *(const v8bf*)(Arow + k1 + 8 + aoff));
        v16bf b1b[4];
        #pragma unroll
        for (int t = 0; t < 4; ++t)
            b1b[t] = BF16CAT(*(const v8bf*)(Brow[t] + k1), *(const v8bf*)(Brow[t] + k1 + 8));

        #pragma unroll
        for (int t = 0; t < 4; ++t)
            acc[t] = __builtin_amdgcn_wmma_f32_16x16x32_bf16(
                false, a0b, false, b0b[t], (short)0, acc[t], false, false);

        // prefetch buffer0 <- k0+64 (wrapped on last iteration: in-bounds, redundant)
        const int k2 = (k0 + 64 < Kk) ? (k0 + 64) : 0;
        a0b = BF16CAT(*(const v8bf*)(Arow + k2 + aoff),
                      *(const v8bf*)(Arow + k2 + 8 + aoff));
        #pragma unroll
        for (int t = 0; t < 4; ++t)
            b0b[t] = BF16CAT(*(const v8bf*)(Brow[t] + k2), *(const v8bf*)(Brow[t] + k2 + 8));

        #pragma unroll
        for (int t = 0; t < 4; ++t)
            acc[t] = __builtin_amdgcn_wmma_f32_16x16x32_bf16(
                false, a1b, false, b1b[t], (short)0, acc[t], false, false);
    }

    #pragma unroll
    for (int t = 0; t < 4; ++t) {
        const int n = tn + t * 16 + r;
        float bv = bias ? bias[n] : 0.0f;
        #pragma unroll
        for (int q = 0; q < 8; ++q) {           // C: VGPR q -> row tm + q + hi*8
            float v = acc[t][q] + bv;
            if (act == 1) v = fmaxf(v, 0.0f);
            else if (act == 2) v = fmaxf(__expf(v), 1e-6f);
            C[(size_t)(tm + q + hi * 8) * ldc + n] = v;
        }
    }
}

// ---------------- conversions ----------------
__global__ void pad_cvt_kernel(const float* __restrict__ X, unsigned short* __restrict__ Y,
                               int cols, int colsPad, size_t total)
{
    size_t i = (size_t)blockIdx.x * blockDim.x + threadIdx.x;
    if (i >= total) return;
    int c = (int)(i % colsPad);
    size_t rw = i / colsPad;
    float v = (c < cols) ? X[rw * cols + c] : 0.0f;
    Y[i] = f32_to_bf16(v);
}

// W (Kdim x Ndim, row-major) -> Wt (Ndim x Kpad, row-major) == B^T, zero-padded K
__global__ void wt_cvt_kernel(const float* __restrict__ W, unsigned short* __restrict__ Wt,
                              int Kdim, int Ndim, int Kpad)
{
    size_t i = (size_t)blockIdx.x * blockDim.x + threadIdx.x;
    size_t total = (size_t)Ndim * Kpad;
    if (i >= total) return;
    int n = (int)(i / Kpad);
    int k = (int)(i % Kpad);
    float v = (k < Kdim) ? W[(size_t)k * Ndim + n] : 0.0f;
    Wt[i] = f32_to_bf16(v);
}

__global__ void cvt_bf16_kernel(const float* __restrict__ X, unsigned short* __restrict__ Y,
                                size_t n, int relu)
{
    size_t i = (size_t)blockIdx.x * blockDim.x + threadIdx.x;
    if (i >= n) return;
    float v = X[i];
    if (relu) v = fmaxf(v, 0.0f);
    Y[i] = f32_to_bf16(v);
}

__global__ void gelu_cvt_kernel(const float* __restrict__ X, unsigned short* __restrict__ Y,
                                size_t n)
{
    size_t i = (size_t)blockIdx.x * blockDim.x + threadIdx.x;
    if (i >= n) return;
    float x = X[i];
    float g = 0.5f * x * (1.0f + erff(x * 0.70710678118654752f));
    Y[i] = f32_to_bf16(g);
}

__global__ void add_relu_cvt_kernel(const float* __restrict__ Aa, const float* __restrict__ Bb,
                                    unsigned short* __restrict__ Y, size_t n)
{
    size_t i = (size_t)blockIdx.x * blockDim.x + threadIdx.x;
    if (i >= n) return;
    float v = Aa[i] + Bb[i];
    Y[i] = f32_to_bf16(fmaxf(v, 0.0f));
}

// ---------------- batchnorm ----------------
__global__ __launch_bounds__(256) void bn_stats_kernel(const float* __restrict__ X, int ld,
                                                       int rows, float* __restrict__ mu,
                                                       float* __restrict__ rstd)
{
    int col = blockIdx.x;
    int tid = threadIdx.x;
    __shared__ float s1[256], s2[256];
    float a = 0.0f, b = 0.0f;
    for (int r = tid; r < rows; r += 256) {
        float v = X[(size_t)r * ld + col];
        a += v; b += v * v;
    }
    s1[tid] = a; s2[tid] = b; __syncthreads();
    for (int s = 128; s > 0; s >>= 1) {
        if (tid < s) { s1[tid] += s1[tid + s]; s2[tid] += s2[tid + s]; }
        __syncthreads();
    }
    if (tid == 0) {
        float m = s1[0] / (float)rows;
        float var = s2[0] / (float)rows - m * m;
        mu[col] = m;
        rstd[col] = rsqrtf(var + 1e-5f);
    }
}

__global__ void bn_apply_kernel(const float* __restrict__ X,
                                const float* __restrict__ mu, const float* __restrict__ rstd,
                                const float* __restrict__ g, const float* __restrict__ be,
                                float* __restrict__ Yf, unsigned short* __restrict__ Ybf,
                                size_t total, int cols, int relu)
{
    size_t i = (size_t)blockIdx.x * blockDim.x + threadIdx.x;
    if (i >= total) return;
    int col = (int)(i % cols);
    float v = (X[i] - mu[col]) * rstd[col] * g[col] + be[col];
    if (relu) v = fmaxf(v, 0.0f);
    if (Yf)  Yf[i] = v;
    if (Ybf) Ybf[i] = f32_to_bf16(v);
}

// ---------------- codebook: ||E||^2, dual softmax + argmax ----------------
__global__ void esq_kernel(const float* __restrict__ E, float* __restrict__ Esq, int K, int D)
{
    int k = blockIdx.x * blockDim.x + threadIdx.x;
    if (k >= K) return;
    float s = 0.0f;
    for (int c = 0; c < D; ++c) { float v = E[(size_t)k * D + c]; s += v * v; }
    Esq[k] = s;
}

// one block (256 thr) per row; K=1024 fixed, CBD=64 fixed
__global__ __launch_bounds__(256) void codebook_softmax_kernel(
    const float* __restrict__ dot,   // N x 1024 : h @ E^T
    const float* __restrict__ h,     // N x 64 (normalized)
    const float* __restrict__ Esq,   // 1024
    const float* __restrict__ gum,   // N x 1024
    float* __restrict__ soft_out,    // N x 1024 (d_out segment)
    unsigned short* __restrict__ asg_out, // N x 1024 bf16
    int* __restrict__ idx_out)       // N
{
    const int row = blockIdx.x;
    const int tid = threadIdx.x;
    __shared__ float red[256];
    __shared__ int  redi[256];

    // ||h_row||^2
    float hv = 0.0f;
    if (tid < 64) { float t = h[(size_t)row * 64 + tid]; hv = t * t; }
    red[tid] = hv; __syncthreads();
    for (int s = 128; s > 0; s >>= 1) { if (tid < s) red[tid] += red[tid + s]; __syncthreads(); }
    float hh = red[0]; __syncthreads();

    float t[4], u[4];
    float mt = -3.402823466e38f, mu2 = -3.402823466e38f;
    int mi = 0;
    #pragma unroll
    for (int j = 0; j < 4; ++j) {
        int col = tid + j * 256;
        float l = 2.0f * dot[(size_t)row * 1024 + col] - hh - Esq[col];
        t[j] = l;
        u[j] = l + gum[(size_t)row * 1024 + col];   // TAU == 1
        if (l > mt) { mt = l; mi = col; }           // first-occurrence within thread
        if (u[j] > mu2) mu2 = u[j];
    }
    red[tid] = mt; redi[tid] = mi; __syncthreads();
    for (int s = 128; s > 0; s >>= 1) {
        if (tid < s) {
            float o = red[tid + s]; int oi = redi[tid + s];
            if (o > red[tid] || (o == red[tid] && oi < redi[tid])) { red[tid] = o; redi[tid] = oi; }
        }
        __syncthreads();
    }
    float maxt = red[0]; int amax = redi[0]; __syncthreads();
    red[tid] = mu2; __syncthreads();
    for (int s = 128; s > 0; s >>= 1) { if (tid < s) red[tid] = fmaxf(red[tid], red[tid + s]); __syncthreads(); }
    float maxu = red[0]; __syncthreads();

    float st = 0.0f, su = 0.0f, et[4], eu[4];
    #pragma unroll
    for (int j = 0; j < 4; ++j) {
        et[j] = __expf(t[j] - maxt);
        eu[j] = __expf(u[j] - maxu);
        st += et[j]; su += eu[j];
    }
    red[tid] = st; __syncthreads();
    for (int s = 128; s > 0; s >>= 1) { if (tid < s) red[tid] += red[tid + s]; __syncthreads(); }
    float sumt = red[0]; __syncthreads();
    red[tid] = su; __syncthreads();
    for (int s = 128; s > 0; s >>= 1) { if (tid < s) red[tid] += red[tid + s]; __syncthreads(); }
    float sumu = red[0];

    float rt = 1.0f / sumt, ru = 1.0f / sumu;
    #pragma unroll
    for (int j = 0; j < 4; ++j) {
        int col = tid + j * 256;
        soft_out[(size_t)row * 1024 + col] = et[j] * rt;
        asg_out[(size_t)row * 1024 + col]  = f32_to_bf16(eu[j] * ru);
    }
    if (tid == 0) idx_out[row] = amax;
}

// ---------------- loss ----------------
__global__ void zero1_kernel(float* p) { if (blockIdx.x == 0 && threadIdx.x == 0) p[0] = 0.0f; }

__global__ __launch_bounds__(256) void loss_accum_kernel(
    const float* __restrict__ Z, int ldz, int coff,
    const float* __restrict__ E, const int* __restrict__ idx,
    float* __restrict__ loss, float scale)
{
    __shared__ float red[256];
    int tid = threadIdx.x;
    size_t i = (size_t)blockIdx.x * 256 + tid;    // over N*64 exactly
    int row = (int)(i >> 6), c = (int)(i & 63);
    float zv = Z[(size_t)row * ldz + coff + c];
    float ev = E[(size_t)idx[row] * 64 + c];
    float d = zv - ev;
    red[tid] = d * d; __syncthreads();
    for (int s = 128; s > 0; s >>= 1) { if (tid < s) red[tid] += red[tid + s]; __syncthreads(); }
    if (tid == 0) atomicAdd(loss, red[0] * scale);
}

// ---------------- host ----------------
static inline int ceil_div(long long a, long long b) { return (int)((a + b - 1) / b); }

static void launch_gemm(const void* A, int lda, const void* Bt, int ldbt,
                        const float* bias, float* C, int ldc,
                        int M, int Nn, int Kk, int act, hipStream_t s)
{
    int tiles = (M / 16) * (Nn / 64);
    int blocks = (tiles + 7) / 8;
    gemm_bf16_wmma<<<blocks, 256, 0, s>>>((const __bf16*)A, lda, (const __bf16*)Bt, ldbt,
                                          bias, C, ldc, M, Nn, Kk, act);
}

extern "C" void kernel_launch(void* const* d_in, const int* in_sizes, int n_in,
                              void* d_out, int out_size, void* d_ws, size_t ws_size,
                              hipStream_t stream)
{
    (void)in_sizes; (void)n_in; (void)out_size; (void)ws_size;
    const int N = 16384, DIN = 2000, DINP = 2048, H0 = 1024, H1 = 512;   // K padded to %64
    const int KCB = 1024, CBD = 64, ZD = 128, HH = 256, FF = 1024;

    const float* exp_in  = (const float*)d_in[0];
    const float* gum     = (const float*)d_in[1];
    const float* enc0_W  = (const float*)d_in[2];
    const float* enc0_b  = (const float*)d_in[3];
    const float* enc0_g  = (const float*)d_in[4];
    const float* enc0_be = (const float*)d_in[5];
    const float* enc1_W  = (const float*)d_in[6];
    const float* enc1_b  = (const float*)d_in[7];
    const float* enc1_g  = (const float*)d_in[8];
    const float* enc1_be = (const float*)d_in[9];
    const float* enc2_W  = (const float*)d_in[10];
    const float* enc2_b  = (const float*)d_in[11];
    const float *cbW[4], *cbb[4], *cbg[4], *cbbe[4], *cbE[4];
    for (int i = 0; i < 4; ++i) {
        cbW[i]  = (const float*)d_in[12 + 5 * i];
        cbb[i]  = (const float*)d_in[13 + 5 * i];
        cbg[i]  = (const float*)d_in[14 + 5 * i];
        cbbe[i] = (const float*)d_in[15 + 5 * i];
        cbE[i]  = (const float*)d_in[16 + 5 * i];
    }
    const float* lin_W  = (const float*)d_in[32]; const float* lin_b  = (const float*)d_in[33];
    const float* fc1_W  = (const float*)d_in[34]; const float* fc1_b  = (const float*)d_in[35];
    const float* fc2_W  = (const float*)d_in[36]; const float* fc2_b  = (const float*)d_in[37];
    const float* g1     = (const float*)d_in[38]; const float* b1     = (const float*)d_in[39];
    const float* g2     = (const float*)d_in[40]; const float* b2     = (const float*)d_in[41];
    const float* mean_W = (const float*)d_in[42]; const float* mean_b = (const float*)d_in[43];
    const float* var_W  = (const float*)d_in[44]; const float* var_b  = (const float*)d_in[45];

    float* out = (float*)d_out;
    const size_t soft_sz  = (size_t)N * KCB;
    const size_t zmean_off = 4 * soft_sz;
    const size_t var_off   = zmean_off + (size_t)N * ZD;
    const size_t loss_off  = var_off + (size_t)N * ZD;
    float* loss_ptr = out + loss_off;

    // ---- workspace layout (lifetime-overlapped regions) ----
    char* base = (char*)d_ws;
    size_t cur = 0;
    auto take = [&](size_t bytes) -> void* {
        void* p = base + cur;
        cur = (cur + bytes + 255) & ~(size_t)255;
        return p;
    };

    // region R_A0: A0 (N x 2048 bf16); later aliased by asg, then gelubf (both N x 1024 bf16)
    unsigned short* A0 = (unsigned short*)take((size_t)N * DINP * 2);
    unsigned short* asg = A0;
    unsigned short* gelubf = A0;
    // weights (persist whole call)
    unsigned short* W0t = (unsigned short*)take((size_t)H0 * DINP * 2);
    unsigned short* W1t = (unsigned short*)take((size_t)H1 * H0 * 2);
    unsigned short* W2t = (unsigned short*)take((size_t)H1 * H1 * 2);
    unsigned short *cbWt[4], *Ebf[4], *Etr[4];
    float* Esq[4];
    for (int i = 0; i < 4; ++i) cbWt[i] = (unsigned short*)take((size_t)CBD * H1 * 2);
    for (int i = 0; i < 4; ++i) Ebf[i]  = (unsigned short*)take((size_t)KCB * CBD * 2);
    for (int i = 0; i < 4; ++i) Etr[i]  = (unsigned short*)take((size_t)CBD * KCB * 2);
    for (int i = 0; i < 4; ++i) Esq[i]  = (float*)take((size_t)KCB * 4);
    unsigned short* linWt  = (unsigned short*)take((size_t)HH * HH * 2);
    unsigned short* fc1t   = (unsigned short*)take((size_t)FF * HH * 2);
    unsigned short* fc2t   = (unsigned short*)take((size_t)HH * FF * 2);
    unsigned short* meanWt = (unsigned short*)take((size_t)ZD * HH * 2);
    unsigned short* varWt  = (unsigned short*)take((size_t)ZD * HH * 2);
    // region BIG0: C1 (N x 1024 f32) -> dot (N x 1024) -> Cf1 (N x 1024)
    float* BIG0 = (float*)take((size_t)N * H0 * 4);
    float* C1 = BIG0; float* dotb = BIG0; float* Cf1 = BIG0;
    // region R_A1: A1 (N x 1024 bf16) -> Z (N x 256 f32 @0) + Zbf (@ N*256*4)
    char* RA1 = (char*)take((size_t)N * H0 * 2);
    unsigned short* A1 = (unsigned short*)RA1;
    float* Z = (float*)RA1;
    unsigned short* Zbf = (unsigned short*)(RA1 + (size_t)N * HH * 4);
    // region R_C2: C2 (N x 512 f32) -> emb (N x 256 @0) + Cf2 (@ N*256*4)
    char* RC2 = (char*)take((size_t)N * H1 * 4);
    float* C2 = (float*)RC2;
    float* emb = (float*)RC2;
    float* Cf2 = (float*)(RC2 + (size_t)N * HH * 4);
    // region R_X: X (N x 512 f32) -> h1 (@0) + h2 (@ N*256*4)
    char* RX = (char*)take((size_t)N * H1 * 4);
    float* X = (float*)RX;
    float* h1 = (float*)RX;
    float* h2 = (float*)(RX + (size_t)N * HH * 4);
    // region R_Xbf: Xbf (N x 512 bf16) -> reluembbf (@0) + h1bf (@ N*256*2)
    char* RXbf = (char*)take((size_t)N * H1 * 2);
    unsigned short* Xbf = (unsigned short*)RXbf;
    unsigned short* reluembbf = (unsigned short*)RXbf;
    unsigned short* h1bf = (unsigned short*)(RXbf + (size_t)N * HH * 2);
    // region R_A2: A2 (N x 512 bf16) -> hbuf (N x 64 f32 @0) + hbf + idx
    char* RA2 = (char*)take((size_t)N * H1 * 2);
    unsigned short* A2 = (unsigned short*)RA2;
    float* hbuf = (float*)RA2;
    unsigned short* hbf = (unsigned short*)(RA2 + (size_t)N * CBD * 4);
    int* idxb = (int*)(RA2 + (size_t)N * CBD * 4 + (size_t)N * CBD * 2);
    float* mu   = (float*)take(1024 * 4);
    float* rstd = (float*)take(1024 * 4);

    // ---- weight / input conversions (every call; deterministic) ----
    {
        size_t tot = (size_t)N * DINP;
        pad_cvt_kernel<<<ceil_div(tot, 256), 256, 0, stream>>>(exp_in, A0, DIN, DINP, tot);
    }
    wt_cvt_kernel<<<ceil_div((size_t)H0 * DINP, 256), 256, 0, stream>>>(enc0_W, W0t, DIN, H0, DINP);
    wt_cvt_kernel<<<ceil_div((size_t)H1 * H0, 256), 256, 0, stream>>>(enc1_W, W1t, H0, H1, H0);
    wt_cvt_kernel<<<ceil_div((size_t)H1 * H1, 256), 256, 0, stream>>>(enc2_W, W2t, H1, H1, H1);
    for (int i = 0; i < 4; ++i) {
        wt_cvt_kernel<<<ceil_div((size_t)CBD * H1, 256), 256, 0, stream>>>(cbW[i], cbWt[i], H1, CBD, H1);
        cvt_bf16_kernel<<<ceil_div((size_t)KCB * CBD, 256), 256, 0, stream>>>(cbE[i], Ebf[i], (size_t)KCB * CBD, 0);
        wt_cvt_kernel<<<ceil_div((size_t)CBD * KCB, 256), 256, 0, stream>>>(cbE[i], Etr[i], KCB, CBD, KCB);
        esq_kernel<<<ceil_div(KCB, 256), 256, 0, stream>>>(cbE[i], Esq[i], KCB, CBD);
    }
    wt_cvt_kernel<<<ceil_div((size_t)HH * HH, 256), 256, 0, stream>>>(lin_W, linWt, HH, HH, HH);
    wt_cvt_kernel<<<ceil_div((size_t)FF * HH, 256), 256, 0, stream>>>(fc1_W, fc1t, HH, FF, HH);
    wt_cvt_kernel<<<ceil_div((size_t)HH * FF, 256), 256, 0, stream>>>(fc2_W, fc2t, FF, HH, FF);
    wt_cvt_kernel<<<ceil_div((size_t)ZD * HH, 256), 256, 0, stream>>>(mean_W, meanWt, HH, ZD, HH);
    wt_cvt_kernel<<<ceil_div((size_t)ZD * HH, 256), 256, 0, stream>>>(var_W, varWt, HH, ZD, HH);

    // ---- encoder ----
    launch_gemm(A0, DINP, W0t, DINP, enc0_b, C1, H0, N, H0, DINP, 0, stream);
    bn_stats_kernel<<<H0, 256, 0, stream>>>(C1, H0, N, mu, rstd);
    bn_apply_kernel<<<ceil_div((size_t)N * H0, 256), 256, 0, stream>>>(
        C1, mu, rstd, enc0_g, enc0_be, nullptr, A1, (size_t)N * H0, H0, 1);
    launch_gemm(A1, H0, W1t, H0, enc1_b, C2, H1, N, H1, H0, 0, stream);
    bn_stats_kernel<<<H1, 256, 0, stream>>>(C2, H1, N, mu, rstd);
    bn_apply_kernel<<<ceil_div((size_t)N * H1, 256), 256, 0, stream>>>(
        C2, mu, rstd, enc1_g, enc1_be, nullptr, A2, (size_t)N * H1, H1, 1);
    launch_gemm(A2, H1, W2t, H1, enc2_b, X, H1, N, H1, H1, 0, stream);
    cvt_bf16_kernel<<<ceil_div((size_t)N * H1, 256), 256, 0, stream>>>(X, Xbf, (size_t)N * H1, 0);

    // ---- codebooks ----
    zero1_kernel<<<1, 1, 0, stream>>>(loss_ptr);
    const float lscale = 1.25f / (float)((size_t)N * CBD);   // (cb + BETA*cm) * SCALER, means identical fwd
    for (int i = 0; i < 4; ++i) {
        launch_gemm(Xbf, H1, cbWt[i], H1, cbb[i], hbuf, CBD, N, CBD, H1, 0, stream);
        bn_stats_kernel<<<CBD, 256, 0, stream>>>(hbuf, CBD, N, mu, rstd);
        bn_apply_kernel<<<ceil_div((size_t)N * CBD, 256), 256, 0, stream>>>(
            hbuf, mu, rstd, cbg[i], cbbe[i], hbuf, hbf, (size_t)N * CBD, CBD, 0);
        // dot = h @ E^T  (Bt == E, 1024x64)
        launch_gemm(hbf, CBD, Ebf[i], CBD, nullptr, dotb, KCB, N, KCB, CBD, 0, stream);
        codebook_softmax_kernel<<<N, 256, 0, stream>>>(
            dotb, hbuf, Esq[i], gum + (size_t)i * N * KCB,
            out + (size_t)i * soft_sz, asg, idxb);
        // z = assignments @ E  (Bt == E^T, 64x1024); write into Z column block i
        launch_gemm(asg, KCB, Etr[i], KCB, nullptr, Z + (size_t)CBD * i, 4 * CBD, N, CBD, KCB, 0, stream);
        loss_accum_kernel<<<(N * CBD) / 256, 256, 0, stream>>>(
            Z, 4 * CBD, CBD * i, cbE[i], idxb, loss_ptr, lscale);
    }

    // ---- decoder head ----
    cvt_bf16_kernel<<<ceil_div((size_t)N * HH, 256), 256, 0, stream>>>(Z, Zbf, (size_t)N * HH, 0);
    launch_gemm(Zbf, HH, linWt, HH, lin_b, emb, HH, N, HH, HH, 1, stream);    // relu
    bn_stats_kernel<<<HH, 256, 0, stream>>>(emb, HH, N, mu, rstd);
    bn_apply_kernel<<<ceil_div((size_t)N * HH, 256), 256, 0, stream>>>(
        emb, mu, rstd, g1, b1, h1, h1bf, (size_t)N * HH, HH, 0);
    launch_gemm(h1bf, HH, fc1t, HH, fc1_b, Cf1, FF, N, FF, HH, 0, stream);
    gelu_cvt_kernel<<<ceil_div((size_t)N * FF, 256), 256, 0, stream>>>(Cf1, gelubf, (size_t)N * FF);
    launch_gemm(gelubf, FF, fc2t, FF, fc2_b, Cf2, HH, N, HH, FF, 0, stream);
    bn_stats_kernel<<<HH, 256, 0, stream>>>(Cf2, HH, N, mu, rstd);
    bn_apply_kernel<<<ceil_div((size_t)N * HH, 256), 256, 0, stream>>>(
        Cf2, mu, rstd, g2, b2, h2, nullptr, (size_t)N * HH, HH, 0);
    add_relu_cvt_kernel<<<ceil_div((size_t)N * HH, 256), 256, 0, stream>>>(
        h1, h2, reluembbf, (size_t)N * HH);
    launch_gemm(reluembbf, HH, meanWt, HH, mean_b, out + zmean_off, ZD, N, ZD, HH, 0, stream);
    launch_gemm(reluembbf, HH, varWt, HH, var_b, out + var_off, ZD, N, ZD, HH, 2, stream); // exp+clamp
}
